// TriplaneSynthesisLayer_61753039781906
// MI455X (gfx1250) — matrix operand order repaired and checked by reading, not compiled, not run on hardware
//
#include <hip/hip_runtime.h>

typedef _Float16 h2   __attribute__((ext_vector_type(2)));
typedef _Float16 v8h  __attribute__((ext_vector_type(8)));
typedef _Float16 v16h __attribute__((ext_vector_type(16)));
typedef float    v4f  __attribute__((ext_vector_type(4)));
typedef float    v8f  __attribute__((ext_vector_type(8)));

#define RDIM 256
#define CDIM 64
#define ODIM 64
#define WDIM 512
#define CC3  192          // 3*C input channels
#define NKT0 18           // group0 K-chunks: 9 taps * 2 channel-halves
#define TSTRIDE 40        // LDS tile row stride in halfs (80B, conflict-free)

// ---------------------------------------------------------------------------
// styles[b][cc] = (w[b] . affine_w[cc]) / sqrt(512) + affine_b[cc]
// ---------------------------------------------------------------------------
__global__ void k_styles(const float* __restrict__ w, const float* __restrict__ aw,
                         const float* __restrict__ ab, float* __restrict__ styles) {
    int idx = blockIdx.x * blockDim.x + threadIdx.x;
    if (idx >= 4 * CC3) return;
    int b = idx / CC3, cc = idx % CC3;
    const float scale = 0.04419417382415922f;   // 1/sqrt(512)
    const float* wp = w + (size_t)b * WDIM;
    const float* ap = aw + (size_t)cc * WDIM;
    float acc = 0.f;
    for (int k = 0; k < WDIM; ++k) acc += wp[k] * ap[k];
    styles[idx] = acc * scale + ab[cc];
}

// ---------------------------------------------------------------------------
// Row means (fn of row r) and col means (fn of column x) per (p,b,c).
// Layout [p][b][c][256]
// ---------------------------------------------------------------------------
__global__ void k_means(const float* __restrict__ x,
                        float* __restrict__ rowmean, float* __restrict__ colmean) {
    int pc = blockIdx.x;                 // 0..767
    int p = pc / 256; int rem = pc % 256;
    int bb = rem / 64; int c = rem % 64;
    const float* plane = x + ((((size_t)bb * 3 + p) * CDIM + c) << 16);
    int t = threadIdx.x;                 // 256 threads
    float s = 0.f;
    for (int r = 0; r < RDIM; ++r) s += plane[r * RDIM + t];
    colmean[(((size_t)p * 4 + bb) * CDIM + c) * RDIM + t] = s * (1.f / 256.f);
    int wv = t >> 5, lane = t & 31;
    for (int r = wv; r < RDIM; r += 8) {
        const float* row = plane + (size_t)r * RDIM;
        float rs = 0.f;
        for (int e = 0; e < 8; ++e) rs += row[lane + 32 * e];
        for (int off = 16; off > 0; off >>= 1) rs += __shfl_down(rs, off, 32);
        if (lane == 0)
            rowmean[(((size_t)p * 4 + bb) * CDIM + c) * RDIM + r] = rs * (1.f / 256.f);
    }
}

// ---------------------------------------------------------------------------
// dcoef[b][o] = rsqrt( sum_cc (sum_tap weight^2) * styles^2 + 1e-8 )
// ---------------------------------------------------------------------------
__global__ void k_dcoef(const float* __restrict__ weight, const float* __restrict__ styles,
                        float* __restrict__ dcoef) {
    int t = threadIdx.x;                 // 256 = 4 batches * 64 outs
    int b = t >> 6, o = t & 63;
    const float* wp = weight + (size_t)o * CC3 * 9;
    float acc = 0.f;
    for (int cc = 0; cc < CC3; ++cc) {
        float w2 = 0.f;
        for (int r = 0; r < 9; ++r) { float v = wp[cc * 9 + r]; w2 += v * v; }
        float s = styles[b * CC3 + cc];
        acc += w2 * s * s;
    }
    dcoef[t] = rsqrtf(acc + 1e-8f);
}

// ---------------------------------------------------------------------------
// Pack styled group0 weights into WMMA A-fragment order (f16), TAP-MAJOR K:
// chunk kt = tap*2 + chalf covers channels cc = chalf*32 + K at tap (ky,kx).
// apre layout: [b][kt(18)][mtile(4)][lane(32)][16 halfs].
// ISA 16-bit A 16x32: half h (v=h>>1,p=h&1):
//   K = (v>=4?16:0) + (lane>=16?8:0) + (v&3)*2 + p ;  M = m*16 + (lane&15)
// ---------------------------------------------------------------------------
__global__ void k_pack(const float* __restrict__ weight, const float* __restrict__ styles,
                       _Float16* __restrict__ apre) {
    int blk = blockIdx.x;                // 4 * 18
    int b = blk / NKT0, kt = blk % NKT0;
    int tap = kt >> 1, chalf = kt & 1;
    int tid = threadIdx.x;
    int m    = tid >> 6;
    int lane = (tid >> 1) & 31;
    int h0   = (tid & 1) * 8;
    int M = m * 16 + (lane & 15);
    _Float16* dst = apre + ((((size_t)b * NKT0 + kt) * 4 + m) * 32 + lane) * 16 + h0;
    for (int q = 0; q < 8; ++q) {
        int h = h0 + q;
        int v = h >> 1, pbit = h & 1;
        int K = ((v >= 4) ? 16 : 0) + ((lane >= 16) ? 8 : 0) + (v & 3) * 2 + pbit;
        int cc = chalf * 32 + K;         // group0 channel
        float val = weight[((size_t)M * CC3 + cc) * 9 + tap] * styles[b * CC3 + cc];
        dst[q] = (_Float16)val;
    }
}

// ---------------------------------------------------------------------------
// Low-rank (broadcast-channel) contributions, exact in f32:
//   G1[n][ky][j][o] = sum_{c,kx} W'[o,64+c,ky,kx]  * 1{j+kx-1 ok} * rm[q1][b][c][j+kx-1]
//   G2[n][kx][i][o] = sum_{c,ky} W'[o,128+c,ky,kx] * 1{i+ky-1 ok} * cm[q2][b][c][i+ky-1]
// ---------------------------------------------------------------------------
__global__ void k_lowrank(const float* __restrict__ weight, const float* __restrict__ styles,
                          const float* __restrict__ rowmean, const float* __restrict__ colmean,
                          float* __restrict__ G1, float* __restrict__ G2) {
    int blk = blockIdx.x;                // 12 * 3 * 256
    int n   = blk / 768;
    int rem = blk - n * 768;
    int t3  = rem >> 8;                  // ky for G1, kx for G2
    int pos = rem & 255;                 // j for G1,  i for G2
    int o   = threadIdx.x;               // 64
    int p = n >> 2, b = n & 3;
    int q1 = p + 2; if (q1 >= 3) q1 -= 3;
    int q2 = p + 1; if (q2 >= 3) q2 -= 3;
    const float* rm = rowmean + ((((size_t)q1 * 4 + b) * CDIM) << 8);
    const float* cm = colmean + ((((size_t)q2 * 4 + b) * CDIM) << 8);
    float a1 = 0.f, a2 = 0.f;
    for (int c = 0; c < CDIM; ++c) {
        float s1 = styles[b * CC3 + 64 + c];
        float s2 = styles[b * CC3 + 128 + c];
        const float* w1 = weight + ((size_t)o * CC3 + 64 + c) * 9 + t3 * 3;  // kx-major
        const float* w2 = weight + ((size_t)o * CC3 + 128 + c) * 9 + t3;     // ky stride 3
        const float* rmc = rm + (c << 8);
        const float* cmc = cm + (c << 8);
        #pragma unroll
        for (int e = 0; e < 3; ++e) {
            int zz = pos + e - 1;
            bool ok = ((unsigned)zz < 256u);
            int zc = ok ? zz : 0;
            a1 += w1[e] * s1 * (ok ? rmc[zc] : 0.f);
            a2 += w2[e * 3] * s2 * (ok ? cmc[zc] : 0.f);
        }
    }
    size_t off = (((size_t)n * 3 + t3) * 256 + pos) * 64 + o;
    G1[off] = a1;
    G2[off] = a2;
}

// ---------------------------------------------------------------------------
// Main kernel helpers: raw-row tile with halo, im2col shift applied at
// LDS-read time. One row-group = (ky, chalf) feeds 3 kx-chunks.
// ---------------------------------------------------------------------------
__device__ __forceinline__ void gatherRow(const float* __restrict__ rp, bool yok,
                                          bool haloT, int haloOff, bool haloZ,
                                          float gv[16], float hv[2]) {
    if (yok) {
        const v4f* p0 = (const v4f*)rp;             // 32B-aligned
        const v4f* p1 = (const v4f*)(rp + 65536);   // next channel
        v4f a0 = p0[0], a1 = p0[1];
        v4f b0 = p1[0], b1 = p1[1];
        #pragma unroll
        for (int e = 0; e < 4; ++e) {
            gv[e]      = a0[e]; gv[4 + e]  = a1[e];
            gv[8 + e]  = b0[e]; gv[12 + e] = b1[e];
        }
        if (haloT) {
            int ho = haloZ ? 0 : haloOff;           // never deref rp[-1]
            float h0 = rp[ho], h1 = rp[65536 + ho];
            hv[0] = haloZ ? 0.f : h0;
            hv[1] = haloZ ? 0.f : h1;
        }
    } else {
        #pragma unroll
        for (int e = 0; e < 16; ++e) gv[e] = 0.f;
        hv[0] = 0.f; hv[1] = 0.f;
    }
}

__device__ __forceinline__ void storeRow(_Float16* __restrict__ buf, int pix0, int kpair,
                                         bool haloT, int haloIdx,
                                         const float gv[16], const float hv[2]) {
    #pragma unroll
    for (int q = 0; q < 8; ++q) {
        h2 t; t[0] = (_Float16)gv[q]; t[1] = (_Float16)gv[8 + q];
        *(h2*)&buf[(pix0 + 1 + q) * TSTRIDE + kpair * 2] = t;   // t-col = pix+1
    }
    if (haloT) {
        h2 t; t[0] = (_Float16)hv[0]; t[1] = (_Float16)hv[1];
        *(h2*)&buf[haloIdx * TSTRIDE + kpair * 2] = t;
    }
}

// ---------------------------------------------------------------------------
// Main implicit-GEMM conv over group0 (K=576, tap-major): one block =
// (image n, row i, 128-pixel half-row). 8 waves x 16 pixels; 4 accumulators
// per wave. Raw-row LDS tile (130 cols, halo) double-buffered; 6 row-groups,
// 3 kx-chunks (12 WMMAs) consumed per tile; one barrier per row-group.
// ---------------------------------------------------------------------------
__global__ void __launch_bounds__(256, 2)
k_main(const float* __restrict__ x,
       const float* __restrict__ bias,
       const float* __restrict__ noise_const,
       const float* __restrict__ noise_strength,
       const float* __restrict__ dcoef,
       const _Float16* __restrict__ apre,
       const float* __restrict__ G1,
       const float* __restrict__ G2,
       float* __restrict__ out) {
    __shared__ _Float16 Blds[2][130 * TSTRIDE];  // [col 0..129][32 ch-halfs]

    int bx  = blockIdx.x;
    int n   = bx >> 9;                       // image 0..11
    int blk = bx & 511;
    int p = n >> 2, b = n & 3;               // plane, batch
    int i  = blk >> 1;                       // output row
    int j0 = (blk & 1) << 7;                 // 0 or 128

    int tid  = threadIdx.x;
    int wv   = tid >> 5, lane = tid & 31;
    int kpair = tid >> 4;                    // 0..15 : channel-pair in tile
    int pixgrp = tid & 15;
    int pix0 = pixgrp * 8;                   // 8 consecutive pixels per thread
    int jbase = j0 + pix0;

    bool haloT = (pixgrp == 0) || (pixgrp == 15);
    int  haloOff = (pixgrp == 0) ? -1 : 8;                    // rel. to jbase
    int  haloIdx = (pixgrp == 0) ? 0 : 129;                   // tile column
    bool haloZ = (pixgrp == 0) ? (j0 == 0) : (j0 == 128);     // outside image

    // per-thread base: image plane + channel-pair + pixel column
    const float* tbase = x + (((size_t)b * 3 + p) * CDIM << 16)
                           + ((size_t)(kpair * 2) << 16) + jbase;

    v8f acc[4] = {};
    float gv[16], hv[2];

    // row-group decode (uniform): rg = ky*2 + chalf
    auto rg_off = [&](int rg, bool& yok) -> int {
        int ky = rg >> 1, chalf = rg & 1;
        int yy = i + ky - 1;
        yok = ((unsigned)yy < 256u);
        int yyc = yy < 0 ? 0 : (yy > 255 ? 255 : yy);
        return (chalf << 21) + (yyc << 8);   // floats
    };

    // prologue: build row-group 0
    {
        bool yok; int soff = rg_off(0, yok);
        gatherRow(tbase + soff, yok, haloT, haloOff, haloZ, gv, hv);
        storeRow(Blds[0], pix0, kpair, haloT, haloIdx, gv, hv);
    }
    __syncthreads();

    _Float16* bcur = Blds[0];
    _Float16* bnxt = Blds[1];

    for (int rg = 0; rg < 6; ++rg) {
        bool more = (rg + 1 < 6);            // uniform scalar branch
        // 1) gather next row-group before the WMMAs (overlaps load latency)
        if (more) {
            bool yok; int soff = rg_off(rg + 1, yok);
            gatherRow(tbase + soff, yok, haloT, haloOff, haloZ, gv, hv);
        }

        // 2) consume 3 kx-chunks from current tile: shift = +kx columns
        int ky = rg >> 1, chalf = rg & 1;
        #pragma unroll
        for (int kx = 0; kx < 3; ++kx) {
            int kt = ((ky * 3 + kx) << 1) + chalf;
            const _Float16* bp = &bcur[(wv * 16 + (lane & 15) + kx) * TSTRIDE
                                       + ((lane >> 4) << 3)];
            v8h bl = *(const v8h*)bp;
            v8h bh = *(const v8h*)(bp + 16);
            v16h bf = __builtin_shufflevector(bl, bh, 0,1,2,3,4,5,6,7,8,9,10,11,12,13,14,15);
            const _Float16* abase = apre + ((((size_t)b * NKT0 + kt) * 4) * 32 + lane) * 16;
            #pragma unroll
            for (int m = 0; m < 4; ++m) {
                const _Float16* ap = abase + (size_t)m * 512;   // 32 lanes * 16 halfs
                v8h al = *(const v8h*)ap;
                v8h ah = *(const v8h*)(ap + 8);
                v16h af = __builtin_shufflevector(al, ah,
                            0,1,2,3,4,5,6,7,8,9,10,11,12,13,14,15);
                acc[m] = __builtin_amdgcn_wmma_f32_16x16x32_f16(
                    false, af, false, bf, (short)0, acc[m], false, false);
            }
        }

        // 3) store next tile into the other buffer, single barrier
        if (more) storeRow(bnxt, pix0, kpair, haloT, haloIdx, gv, hv);
        _Float16* tswap = bcur; bcur = bnxt; bnxt = tswap;
        __syncthreads();
    }

    // ---- epilogue: + low-rank terms, dcoef, noise, bias, lrelu*sqrt2, clamp --
    int jj = j0 + wv * 16 + (lane & 15);
    float nz = noise_const[i * RDIM + jj] * noise_strength[0];
    int hi8 = (lane >> 4) * 8;
    bool iok0 = (i > 0), iok2 = (i < 255);
    bool jok0 = (jj > 0), jok2 = (jj < 255);
    const float* g1p = G1 + (((size_t)n * 3) * 256 + jj) * 64;   // +ky*16384 +o
    const float* g2p = G2 + (((size_t)n * 3) * 256 + i) * 64;    // +kx*16384 +o
    float* ob = out + ((((size_t)b * 3 + p) * ODIM) * RDIM + i) * RDIM + jj;
    #pragma unroll
    for (int m = 0; m < 4; ++m) {
        #pragma unroll
        for (int r = 0; r < 8; ++r) {
            int o = m * 16 + hi8 + r;
            float g1 = g1p[16384 + o]
                     + (iok0 ? g1p[o] : 0.f)
                     + (iok2 ? g1p[32768 + o] : 0.f);
            float g2 = g2p[16384 + o]
                     + (jok0 ? g2p[o] : 0.f)
                     + (jok2 ? g2p[32768 + o] : 0.f);
            float v = (acc[m][r] + g1 + g2) * dcoef[b * ODIM + o] + nz + bias[o];
            v = (v > 0.f) ? v : 0.2f * v;
            v *= 1.4142135623730951f;
            v = fminf(256.f, fmaxf(-256.f, v));
            ob[(size_t)o << 16] = v;
        }
    }
}

// ---------------------------------------------------------------------------
extern "C" void kernel_launch(void* const* d_in, const int* in_sizes, int n_in,
                              void* d_out, int out_size, void* d_ws, size_t ws_size,
                              hipStream_t stream) {
    const float* x              = (const float*)d_in[0];
    const float* w              = (const float*)d_in[1];
    const float* affine_w       = (const float*)d_in[2];
    const float* affine_b       = (const float*)d_in[3];
    const float* weight         = (const float*)d_in[4];
    const float* bias           = (const float*)d_in[5];
    const float* noise_const    = (const float*)d_in[6];
    const float* noise_strength = (const float*)d_in[7];
    float* out = (float*)d_out;

    char* ws = (char*)d_ws;
    float*    styles  = (float*)(ws);                        //   768 f32
    float*    dcoef   = (float*)(ws + 4096);                 //   256 f32
    float*    rowmean = (float*)(ws + 8192);                 // 196608 f32
    float*    colmean = (float*)(ws + 8192 + 786432);        // 196608 f32
    _Float16* apre    = (_Float16*)(ws + 8192 + 2*786432);   // 147456 f16 (288KB)
    float*    G1      = (float*)(ws + 8192 + 2*786432 + 294912);           // 2.25MB
    float*    G2      = (float*)(ws + 8192 + 2*786432 + 294912 + 2359296); // 2.25MB

    hipLaunchKernelGGL(k_styles,  dim3(3),    dim3(256), 0, stream, w, affine_w, affine_b, styles);
    hipLaunchKernelGGL(k_means,   dim3(768),  dim3(256), 0, stream, x, rowmean, colmean);
    hipLaunchKernelGGL(k_dcoef,   dim3(1),    dim3(256), 0, stream, weight, styles, dcoef);
    hipLaunchKernelGGL(k_pack,    dim3(72),   dim3(256), 0, stream, weight, styles, apre);
    hipLaunchKernelGGL(k_lowrank, dim3(9216), dim3(64),  0, stream,
                       weight, styles, rowmean, colmean, G1, G2);
    hipLaunchKernelGGL(k_main,    dim3(6144), dim3(256), 0, stream,
                       x, bias, noise_const, noise_strength, dcoef, apre, G1, G2, out);
}
